// BiLSTMCRF_56642028699771
// MI455X (gfx1250) — matrix operand order repaired and checked by reading, not compiled, not run on hardware
//
#include <hip/hip_runtime.h>
#include <hip/hip_bf16.h>

// ---------------------------------------------------------------------------
// BiLSTM-CRF inference for MI455X (gfx1250, wave32)
//   L=2048, E=256, H=256 (per dir), 4H=1024 gates, HID=512, T_TAGS=6
//   - batched input projections: bf16 WMMA (v_wmma_f32_16x16x32_bf16)
//   - sequential recurrence: fp32 VALU matvec, 1 workgroup per direction
//   - viterbi: single wave
// ---------------------------------------------------------------------------

#define LSEQ   2048
#define EDIM   256
#define HDIM   256
#define GDIM   1024   // 4*H
#define HID2   512    // 2*H
#define NTAGS  6
#define START_TAG 4
#define STOP_TAG  5
#define NEGINF (-10000.0f)

typedef __attribute__((ext_vector_type(16))) __bf16 bf16x16;
typedef __attribute__((ext_vector_type(8)))  float  floatx8;

union FragBF {
    bf16x16 v;
    uint4   u[2];
};

static __device__ __forceinline__ unsigned short f32_to_bf16_rne(float f) {
    unsigned int u = __float_as_uint(f);
    unsigned int r = (u + 0x7FFFu + ((u >> 16) & 1u)) >> 16;
    return (unsigned short)r;
}

// ------------------------- embedding gather (f32 -> bf16) -------------------
__global__ void embed_gather_bf16(const int* __restrict__ x,
                                  const float* __restrict__ emb,
                                  unsigned short* __restrict__ out,
                                  int L, int E) {
    int i = blockIdx.x * blockDim.x + threadIdx.x;
    if (i >= L * E) return;
    int l = i / E, e = i - l * E;
    out[i] = f32_to_bf16_rne(emb[(size_t)x[l] * E + e]);
}

// ------------------------- generic f32 -> bf16 convert ----------------------
__global__ void cvt_f32_bf16(const float* __restrict__ in,
                             unsigned short* __restrict__ out, int n) {
    int i = blockIdx.x * blockDim.x + threadIdx.x;
    if (i >= n) return;
    out[i] = f32_to_bf16_rne(in[i]);
}

// ------------------------- WMMA GEMM:  OUT[M][N] = A[M][K] * W[N][K]^T + b --
// A, W in bf16 (K contiguous).  One wave per 16x16 output tile.
// Fragment layouts follow CDNA5 ISA 16-bit A (16x32) / B (32x16) / f32 C (16x16).
__global__ void gemm_bf16_wmma(const unsigned short* __restrict__ A,
                               const unsigned short* __restrict__ W,
                               const float* __restrict__ bias1,
                               const float* __restrict__ bias2,
                               float* __restrict__ OUT,
                               int M, int N, int K) {
    int wave  = (blockIdx.x * blockDim.x + threadIdx.x) >> 5;
    int lane  = threadIdx.x & 31;
    int tilesN = N >> 4;
    int tilesM = M >> 4;
    if (wave >= tilesM * tilesN) return;
    int m0 = (wave / tilesN) << 4;
    int n0 = (wave - (wave / tilesN) * tilesN) << 4;

    int  ml = lane & 15;
    bool hi = lane >= 16;

    // A: lane holds row (m0+ml); halves 0..7 at K = k0 + (hi?8:0), halves 8..15 at K+16
    const unsigned short* arow = A + (size_t)(m0 + ml) * K + (hi ? 8 : 0);
    // B: lane holds column (n0+ml); halves 0..7 at K = k0 + (hi?8:0), halves 8..15 at K+16
    const unsigned short* brow = W + (size_t)(n0 + ml) * K + (hi ? 8 : 0);

    floatx8 acc = {};
    for (int k0 = 0; k0 < K; k0 += 32) {
        FragBF a, b;
        a.u[0] = *(const uint4*)(arow + k0);        // K = k0+c .. k0+c+7
        a.u[1] = *(const uint4*)(arow + k0 + 16);   // K = k0+16+c .. +7
        b.u[0] = *(const uint4*)(brow + k0);        // K = k0+c .. k0+c+7
        b.u[1] = *(const uint4*)(brow + k0 + 16);   // K = k0+16+c .. +7
        acc = __builtin_amdgcn_wmma_f32_16x16x32_bf16(
            false, a.v, false, b.v, (short)0, acc, false, false);
    }

    int   n  = n0 + ml;
    float bb = 0.0f;
    if (bias1) bb += bias1[n];
    if (bias2) bb += bias2[n];
    int mb = m0 + (hi ? 8 : 0);   // C layout: VGPR r -> row mb + r
    #pragma unroll
    for (int r = 0; r < 8; ++r)
        OUT[(size_t)(mb + r) * N + n] = acc[r] + bb;
}

// ------------------------- dual-direction LSTM recurrence -------------------
// gridDim.x == 2 : block 0 = forward, block 1 = reverse.
// Zin already holds  W_ih@x + b_ih + b_hh  for every timestep.
// out[t][outOff + g] = h_t  (out stride = 512, fwd cols 0..255, rev 256..511)
__global__ void __launch_bounds__(1024)
lstm_recur2(const float* __restrict__ z_f, const float* __restrict__ z_r,
            const float* __restrict__ whh_f, const float* __restrict__ whh_r,
            const float* __restrict__ h0, const float* __restrict__ c0,
            int h0row_f, int h0row_r,
            float* __restrict__ out, int L) {
    const int dir = blockIdx.x;
    const float* Z    = dir ? z_r   : z_f;
    const float* Whh  = dir ? whh_r : whh_f;
    const int    row0 = dir ? h0row_r : h0row_f;
    const int    oOff = dir * HDIM;
    const int    g    = threadIdx.x;            // 0..1023

    __shared__ float h_s[HDIM];
    __shared__ float z_s[GDIM];

    float c = 0.0f;
    if (g < HDIM) {
        h_s[g] = h0[(size_t)row0 * HDIM + g];
        c      = c0[(size_t)row0 * HDIM + g];
    }
    __syncthreads();

    const float* wrow = Whh + (size_t)g * HDIM;
    for (int s = 0; s < L; ++s) {
        const int t = dir ? (L - 1 - s) : s;
        float dot = 0.0f;
        #pragma unroll 8
        for (int k = 0; k < HDIM; ++k) dot += wrow[k] * h_s[k];
        z_s[g] = Z[(size_t)t * GDIM + g] + dot;
        __syncthreads();
        if (g < HDIM) {
            float iv = 1.0f / (1.0f + __expf(-z_s[g]));
            float fv = 1.0f / (1.0f + __expf(-z_s[HDIM + g]));
            float gv = tanhf(z_s[2 * HDIM + g]);
            float ov = 1.0f / (1.0f + __expf(-z_s[3 * HDIM + g]));
            c = fv * c + iv * gv;
            float hv = ov * tanhf(c);
            h_s[g] = hv;
            out[(size_t)t * HID2 + oOff + g] = hv;
        }
        __syncthreads();
    }
}

// ------------------------- hidden -> tag scores -----------------------------
__global__ void hid2tag_kernel(const float* __restrict__ hid,
                               const float* __restrict__ wtag,
                               const float* __restrict__ btag,
                               float* __restrict__ feats, int L) {
    int i = blockIdx.x * blockDim.x + threadIdx.x;
    if (i >= L * NTAGS) return;
    int l = i / NTAGS, tg = i - l * NTAGS;
    const float* h = hid + (size_t)l * HID2;
    const float* w = wtag + (size_t)tg * HID2;
    float s = btag[tg];
    #pragma unroll 8
    for (int k = 0; k < HID2; ++k) s += h[k] * w[k];
    feats[i] = s;
}

// ------------------------- Viterbi decode (single wave) ---------------------
__global__ void viterbi_kernel(const float* __restrict__ feats,
                               const float* __restrict__ trans,
                               int* __restrict__ bp,
                               int* __restrict__ path, int L) {
    __shared__ float fv[8];
    int t = threadIdx.x;
    if (t < NTAGS) fv[t] = (t == START_TAG) ? 0.0f : NEGINF;
    __syncthreads();
    for (int l = 0; l < L; ++l) {
        float best = -3.4e38f;
        int   bi   = 0;
        if (t < NTAGS) {
            for (int p = 0; p < NTAGS; ++p) {
                float s = fv[p] + trans[t * NTAGS + p];
                if (s > best) { best = s; bi = p; }
            }
        }
        __syncthreads();
        if (t < NTAGS) {
            fv[t] = best + feats[l * NTAGS + t];
            bp[l * NTAGS + t] = bi;
        }
        __syncthreads();
    }
    if (t == 0) {
        float best = -3.4e38f;
        int   tag  = 0;
        for (int p = 0; p < NTAGS; ++p) {
            float s = fv[p] + trans[STOP_TAG * NTAGS + p];
            if (s > best) { best = s; tag = p; }
        }
        for (int l = L - 1; l >= 0; --l) {
            path[l] = tag;
            tag = bp[l * NTAGS + tag];
        }
    }
}

// ---------------------------------------------------------------------------
extern "C" void kernel_launch(void* const* d_in, const int* in_sizes, int n_in,
                              void* d_out, int out_size, void* d_ws, size_t ws_size,
                              hipStream_t stream) {
    const int L = LSEQ;

    const int*   x        = (const int*)  d_in[0];
    const float* emb      = (const float*)d_in[1];
    const float* w_ih_l0  = (const float*)d_in[2];
    const float* w_hh_l0  = (const float*)d_in[3];
    const float* b_ih_l0  = (const float*)d_in[4];
    const float* b_hh_l0  = (const float*)d_in[5];
    const float* w_ih_l0r = (const float*)d_in[6];
    const float* w_hh_l0r = (const float*)d_in[7];
    const float* b_ih_l0r = (const float*)d_in[8];
    const float* b_hh_l0r = (const float*)d_in[9];
    const float* w_ih_l1  = (const float*)d_in[10];
    const float* w_hh_l1  = (const float*)d_in[11];
    const float* b_ih_l1  = (const float*)d_in[12];
    const float* b_hh_l1  = (const float*)d_in[13];
    const float* w_ih_l1r = (const float*)d_in[14];
    const float* w_hh_l1r = (const float*)d_in[15];
    const float* b_ih_l1r = (const float*)d_in[16];
    const float* b_hh_l1r = (const float*)d_in[17];
    const float* w_tag    = (const float*)d_in[18];
    const float* b_tag    = (const float*)d_in[19];
    const float* trans    = (const float*)d_in[20];
    const float* h0       = (const float*)d_in[21];
    const float* c0       = (const float*)d_in[22];
    int*         path     = (int*)d_out;

    // ---- workspace layout (bytes, 256-aligned blocks) ----
    char* ws = (char*)d_ws;
    unsigned short* xs_bf    = (unsigned short*)(ws + 0);          // L*E bf16      (1.0 MB)
    unsigned short* wih0_bf  = (unsigned short*)(ws + 1048576);    // 1024*256 bf16 (0.5 MB)
    unsigned short* wih0r_bf = (unsigned short*)(ws + 1572864);    // 0.5 MB
    unsigned short* wih1_bf  = (unsigned short*)(ws + 2097152);    // 1024*512 bf16 (1 MB)
    unsigned short* wih1r_bf = (unsigned short*)(ws + 3145728);    // 1 MB
    float*          z_a      = (float*)(ws + 4194304);             // L*1024 f32 (8 MB) reused l0f/l1f
    float*          z_b      = (float*)(ws + 12582912);            // 8 MB, reused l0r/l1r
    float*          xs1      = (float*)(ws + 20971520);            // L*512 f32 (4 MB)
    unsigned short* xs1_bf   = (unsigned short*)(ws + 25165824);   // L*512 bf16 (2 MB)
    float*          hid2     = (float*)(ws + 27262976);            // L*512 f32 (4 MB)
    float*          feats    = (float*)(ws + 31457280);            // L*6 f32
    int*            bp       = (int*)  (ws + 31506432);            // L*6 i32

    // ---- stage A: embedding gather + bf16 conversions ----
    {
        int n = L * EDIM;
        embed_gather_bf16<<<(n + 255) / 256, 256, 0, stream>>>(x, emb, xs_bf, L, EDIM);
    }
    cvt_f32_bf16<<<(GDIM * EDIM + 255) / 256, 256, 0, stream>>>(w_ih_l0,  wih0_bf,  GDIM * EDIM);
    cvt_f32_bf16<<<(GDIM * EDIM + 255) / 256, 256, 0, stream>>>(w_ih_l0r, wih0r_bf, GDIM * EDIM);
    cvt_f32_bf16<<<(GDIM * HID2 + 255) / 256, 256, 0, stream>>>(w_ih_l1,  wih1_bf,  GDIM * HID2);
    cvt_f32_bf16<<<(GDIM * HID2 + 255) / 256, 256, 0, stream>>>(w_ih_l1r, wih1r_bf, GDIM * HID2);

    // ---- stage B: layer-0 input projections (WMMA) ----
    {
        int tiles = (L / 16) * (GDIM / 16);             // 8192 waves
        int blocks = (tiles * 32 + 255) / 256;
        gemm_bf16_wmma<<<blocks, 256, 0, stream>>>(xs_bf, wih0_bf,  b_ih_l0,  b_hh_l0,
                                                   z_a, L, GDIM, EDIM);
        gemm_bf16_wmma<<<blocks, 256, 0, stream>>>(xs_bf, wih0r_bf, b_ih_l0r, b_hh_l0r,
                                                   z_b, L, GDIM, EDIM);
    }

    // ---- stage C: layer-0 recurrence (fwd + rev concurrently) ----
    lstm_recur2<<<2, 1024, 0, stream>>>(z_a, z_b, w_hh_l0, w_hh_l0r,
                                        h0, c0, /*h0 rows*/ 0, 1, xs1, L);

    // ---- stage D: layer-1 input projections ----
    cvt_f32_bf16<<<(L * HID2 + 255) / 256, 256, 0, stream>>>(xs1, xs1_bf, L * HID2);
    {
        int tiles = (L / 16) * (GDIM / 16);
        int blocks = (tiles * 32 + 255) / 256;
        gemm_bf16_wmma<<<blocks, 256, 0, stream>>>(xs1_bf, wih1_bf,  b_ih_l1,  b_hh_l1,
                                                   z_a, L, GDIM, HID2);
        gemm_bf16_wmma<<<blocks, 256, 0, stream>>>(xs1_bf, wih1r_bf, b_ih_l1r, b_hh_l1r,
                                                   z_b, L, GDIM, HID2);
    }

    // ---- stage E: layer-1 recurrence ----
    lstm_recur2<<<2, 1024, 0, stream>>>(z_a, z_b, w_hh_l1, w_hh_l1r,
                                        h0, c0, /*h0 rows*/ 2, 3, hid2, L);

    // ---- stage F: tag scores ----
    hid2tag_kernel<<<(L * NTAGS + 255) / 256, 256, 0, stream>>>(hid2, w_tag, b_tag, feats, L);

    // ---- stage G: Viterbi decode ----
    viterbi_kernel<<<1, 32, 0, stream>>>(feats, trans, bp, path, L);
}
